// EdgeConvClassifier_54082228191640
// MI455X (gfx1250) — matrix operand order repaired and checked by reading, not compiled, not run on hardware
//
#include <hip/hip_runtime.h>
#include <hip/hip_bf16.h>
#include <math.h>

// ---------------------------------------------------------------------------
// EdgeConv classifier for MI455X (gfx1250, wave32, WMMA), transposed form:
//   D[n][edge] = W^T @ msg^T  so that B-fragments are per-lane contiguous
//   gathers and the D->B inter-layer conversion is a single lane^16 shuffle.
// All f32->f16 conversion uses packed cvt_pkrtz; lrelu = max(x, slope*x).
// Aggregation: order-preserving int-keyed global_atomic_max_i32.
// ---------------------------------------------------------------------------

typedef __attribute__((ext_vector_type(16))) _Float16 v16h;
typedef __attribute__((ext_vector_type(8)))  _Float16 v8h;
typedef __attribute__((ext_vector_type(2)))  __fp16   f16x2;  // cvt_pkrtz return type
typedef __attribute__((ext_vector_type(8)))  float    v8f;
typedef __attribute__((ext_vector_type(4)))  float    v4f;
typedef __attribute__((ext_vector_type(8)))  unsigned v8u;

#define N_NODES   50000
#define N_EDGES   1280000
#define N_GRAPHS  64
#define IN_CH     6
#define HID       64
#define SLOPE     0.2f
#define WAVES_PB  8
#define KEY_NEG_INF ((int)0x807FFFFF)   // f32key(-inf)

// 0 < slope < 1  =>  leaky_relu(x) == max(x, slope*x)   (2 VALU ops, dual-issue)
__device__ __forceinline__ float lrelu(float x) { return fmaxf(x, SLOPE * x); }
__device__ __forceinline__ int   f32key(float f) { int i = __float_as_int(f); return i >= 0 ? i : (i ^ 0x7FFFFFFF); }
__device__ __forceinline__ float keyf32(int k)   { return __int_as_float(k >= 0 ? k : (k ^ 0x7FFFFFFF)); }

__device__ __forceinline__ unsigned pk2(float a, float b) {
    f16x2 p = __builtin_amdgcn_cvt_pkrtz(a, b);
    return __builtin_bit_cast(unsigned, p);
}

// ---------------------------------------------------------------------------
__global__ void fill_i32(int* p, int n, int v) {
    int i = blockIdx.x * blockDim.x + threadIdx.x;
    if (i < n) p[i] = v;
}
__global__ void fill_f32(float* p, int n, float v) {
    int i = blockIdx.x * blockDim.x + threadIdx.x;
    if (i < n) p[i] = v;
}
__global__ void decode_max(const int* __restrict__ keys, float* __restrict__ h, int n) {
    int i = blockIdx.x * blockDim.x + threadIdx.x;
    if (i < n) {
        int k = keys[i];
        h[i] = (k == KEY_NEG_INF) ? 0.f : keyf32(k);
    }
}

// ---------------------------------------------------------------------------
// Shared tail: H (transposed D-layout f32: rows = hidden k, cols = edges)
// -> pack to f16 pairs -> lane^16 shuffle -> B fragments -> matmul2 (W4^T@H)
// -> lrelu -> keyed atomic segment-max onto dst (column = own edge).
// ---------------------------------------------------------------------------
__device__ __forceinline__ void mlp2_and_aggregate(
    const v8f acc1[4], const v16h aWb[2][4], const v8f biasB[4],
    int dN, bool valid, int hi, int ln, int lane, int* __restrict__ outKeys)
{
    // lane holds H[k = t*16 + 8*hi + r][col=ln]; pack pairs, swap with lane^16
    unsigned hl[4][4], hp[4][4];
    #pragma unroll
    for (int t = 0; t < 4; ++t)
        #pragma unroll
        for (int j = 0; j < 4; ++j) {
            hl[t][j] = pk2(lrelu(acc1[t][2*j]), lrelu(acc1[t][2*j + 1]));
            hp[t][j] = (unsigned)__shfl((int)hl[t][j], lane ^ 16, 32);
        }

    v8f acc2[4];
    #pragma unroll
    for (int t = 0; t < 4; ++t) acc2[t] = biasB[t];

    #pragma unroll
    for (int c = 0; c < 2; ++c) {
        // B frag element e: K = c*32 + 16*hi + e  (u32-granularity selects)
        v8u bu;
        #pragma unroll
        for (int j = 0; j < 4; ++j) bu[j]     = hi ? hp[2*c + 1][j] : hl[2*c][j];
        #pragma unroll
        for (int j = 0; j < 4; ++j) bu[4 + j] = hi ? hl[2*c + 1][j] : hp[2*c][j];
        v16h b2 = __builtin_bit_cast(v16h, bu);
        #pragma unroll
        for (int t = 0; t < 4; ++t)
            acc2[t] = __builtin_amdgcn_wmma_f32_16x16x32_f16(
                false, aWb[c][t], false, b2, (short)0, acc2[t], false, false);
    }

    if (valid) {
        #pragma unroll
        for (int t = 0; t < 4; ++t)
            #pragma unroll
            for (int r = 0; r < 8; ++r) {
                float v = lrelu(acc2[t][r]);           // channel = t*16 + 8*hi + r
                atomicMax(&outKeys[dN * HID + t*16 + 8*hi + r], f32key(v));
            }
    }
}

// ---------------------------------------------------------------------------
// Layer 1: FIN=6 (K padded to 32).  All weight A-fragments in registers.
// ---------------------------------------------------------------------------
__global__ __launch_bounds__(256, 1)
void edgeconv1_wmma(const float* __restrict__ x,
                    const int* __restrict__ src, const int* __restrict__ dst,
                    const float* __restrict__ Wa, const float* __restrict__ ba,
                    const float* __restrict__ Wb, const float* __restrict__ bb,
                    int* __restrict__ outKeys, int nEdges)
{
    const int lane = threadIdx.x & 31;
    const int wave = threadIdx.x >> 5;
    const int hi = lane >> 4, ln = lane & 15;

    // loop-invariant weight fragments (A layout: row = ln, K mapping)
    v16h aWa[4], aWb[2][4];
    #pragma unroll
    for (int t = 0; t < 4; ++t)
        #pragma unroll
        for (int e = 0; e < 16; ++e) {
            int k = e + 16*hi;
            aWa[t][e] = (_Float16)((k < 2*IN_CH) ? Wa[k*HID + t*16 + ln] : 0.f);
        }
    #pragma unroll
    for (int c = 0; c < 2; ++c)
        #pragma unroll
        for (int t = 0; t < 4; ++t)
            #pragma unroll
            for (int e = 0; e < 16; ++e) {
                int k = c*32 + e + 16*hi;
                aWb[c][t][e] = (_Float16)Wb[k*HID + t*16 + ln];
            }
    v8f biasA[4], biasB[4];
    #pragma unroll
    for (int t = 0; t < 4; ++t)
        #pragma unroll
        for (int r = 0; r < 8; ++r) {
            biasA[t][r] = ba[t*16 + 8*hi + r];
            biasB[t][r] = bb[t*16 + 8*hi + r];
        }

    const int nTiles = (nEdges + 15) / 16;
    for (int tile = blockIdx.x * WAVES_PB + wave; tile < nTiles;
         tile += gridDim.x * WAVES_PB)
    {
        const int  myEdge = tile * 16 + ln;
        const bool valid  = (myEdge < nEdges);
        const int  sN     = valid ? src[myEdge] : 0;
        const int  dN     = valid ? dst[myEdge] : 0;

        float xd[IN_CH], xs[IN_CH];
        #pragma unroll
        for (int j = 0; j < IN_CH; ++j) {
            xd[j] = x[dN * IN_CH + j];
            xs[j] = x[sN * IN_CH + j];
        }

        // msg[k = e + 16*hi][edge = ln]; K>=12 zero pad, hi half all zero
        float mv[16];
        #pragma unroll
        for (int e = 0; e < 16; ++e) {
            float v = 0.f;
            if (e < IN_CH)          v = xd[e];
            else if (e < 2*IN_CH)   v = xs[e - IN_CH] - xd[e - IN_CH];
            mv[e] = v;
        }
        v8u bu;
        #pragma unroll
        for (int j = 0; j < 8; ++j)
            bu[j] = hi ? 0u : pk2(mv[2*j], mv[2*j + 1]);
        v16h bm = __builtin_bit_cast(v16h, bu);

        v8f acc1[4];
        #pragma unroll
        for (int t = 0; t < 4; ++t) acc1[t] = biasA[t];
        #pragma unroll
        for (int t = 0; t < 4; ++t)
            acc1[t] = __builtin_amdgcn_wmma_f32_16x16x32_f16(
                false, aWa[t], false, bm, (short)0, acc1[t], false, false);

        mlp2_and_aggregate(acc1, aWb, biasB, dN, valid, hi, ln, lane, outKeys);
    }
}

// ---------------------------------------------------------------------------
// Layer 2: FIN=64 (K=128).  W3^T staged in padded LDS (b128 reads),
// W4^T fragments in registers, message gathers as global_load_b128,
// endpoint rows converted once to packed f16, diff done in packed f16.
// ---------------------------------------------------------------------------
__global__ __launch_bounds__(256, 1)
void edgeconv2_wmma(const float* __restrict__ x,
                    const int* __restrict__ src, const int* __restrict__ dst,
                    const float* __restrict__ Wa, const float* __restrict__ ba,
                    const float* __restrict__ Wb, const float* __restrict__ bb,
                    int* __restrict__ outKeys, int nEdges)
{
    constexpr int KP  = 128;
    constexpr int PAD = 8;                      // halves; breaks 256B bank cycle
    __shared__ _Float16 sWaT[HID][KP + PAD];    // transposed W3: [n][k]

    for (int i = threadIdx.x; i < HID * KP; i += blockDim.x) {
        int n = i % HID, k = i / HID;
        sWaT[n][k] = (_Float16)Wa[k * HID + n];
    }
    __syncthreads();

    const int lane = threadIdx.x & 31;
    const int wave = threadIdx.x >> 5;
    const int hi = lane >> 4, ln = lane & 15;

    v16h aWb[2][4];
    #pragma unroll
    for (int c = 0; c < 2; ++c)
        #pragma unroll
        for (int t = 0; t < 4; ++t)
            #pragma unroll
            for (int e = 0; e < 16; ++e) {
                int k = c*32 + e + 16*hi;
                aWb[c][t][e] = (_Float16)Wb[k*HID + t*16 + ln];
            }
    v8f biasA[4], biasB[4];
    #pragma unroll
    for (int t = 0; t < 4; ++t)
        #pragma unroll
        for (int r = 0; r < 8; ++r) {
            biasA[t][r] = ba[t*16 + 8*hi + r];
            biasB[t][r] = bb[t*16 + 8*hi + r];
        }

    const int nTiles = (nEdges + 15) / 16;
    for (int tile = blockIdx.x * WAVES_PB + wave; tile < nTiles;
         tile += gridDim.x * WAVES_PB)
    {
        const int  myEdge = tile * 16 + ln;
        const bool valid  = (myEdge < nEdges);
        const int  sN     = valid ? src[myEdge] : 0;
        const int  dN     = valid ? dst[myEdge] : 0;

        // lane's edge needs rows at float offsets {16*hi, 32+16*hi} of both
        // endpoints: 8x global_load_b128 per endpoint.
        const v4f* xd0 = (const v4f*)(x + (size_t)dN * HID);
        const v4f* xs0 = (const v4f*)(x + (size_t)sN * HID);
        v4f dseg[2][4], sseg[2][4];
        #pragma unroll
        for (int p = 0; p < 2; ++p)
            #pragma unroll
            for (int q = 0; q < 4; ++q) {
                dseg[p][q] = xd0[p*8 + 4*hi + q];
                sseg[p][q] = xs0[p*8 + 4*hi + q];
            }
        // convert once to packed f16; diff computed as packed-f16 subtract
        f16x2 dpk[2][8], spk[2][8];
        #pragma unroll
        for (int p = 0; p < 2; ++p)
            #pragma unroll
            for (int j = 0; j < 8; ++j) {
                dpk[p][j] = __builtin_amdgcn_cvt_pkrtz(dseg[p][j >> 1][(j & 1)*2],
                                                       dseg[p][j >> 1][(j & 1)*2 + 1]);
                spk[p][j] = __builtin_amdgcn_cvt_pkrtz(sseg[p][j >> 1][(j & 1)*2],
                                                       sseg[p][j >> 1][(j & 1)*2 + 1]);
            }

        v8f acc1[4];
        #pragma unroll
        for (int t = 0; t < 4; ++t) acc1[t] = biasA[t];

        #pragma unroll
        for (int c = 0; c < 4; ++c) {
            // B frag: K = c*32 + 16*hi + e ; msg = [h_d | h_s - h_d]
            v8u bu;
            #pragma unroll
            for (int j = 0; j < 8; ++j) {
                f16x2 pr;
                if      (c == 0) pr = dpk[0][j];
                else if (c == 1) pr = dpk[1][j];
                else if (c == 2) pr = spk[0][j] - dpk[0][j];
                else             pr = spk[1][j] - dpk[1][j];
                bu[j] = __builtin_bit_cast(unsigned, pr);
            }
            v16h bm = __builtin_bit_cast(v16h, bu);
            #pragma unroll
            for (int t = 0; t < 4; ++t) {
                // A frag from LDS: two aligned 16B runs (k = c*32+16*hi+{0..7}, +8)
                const v8h* r0 = (const v8h*)&sWaT[t*16 + ln][c*32 + 16*hi];
                const v8h* r1 = (const v8h*)&sWaT[t*16 + ln][c*32 + 16*hi + 8];
                v16h a = __builtin_shufflevector(r0[0], r1[0],
                    0,1,2,3,4,5,6,7,8,9,10,11,12,13,14,15);
                acc1[t] = __builtin_amdgcn_wmma_f32_16x16x32_f16(
                    false, a, false, bm, (short)0, acc1[t], false, false);
            }
        }

        mlp2_and_aggregate(acc1, aWb, biasB, dN, valid, hi, ln, lane, outKeys);
    }
}

// ---------------------------------------------------------------------------
__global__ void pool_kernel(const float* __restrict__ h, const int* __restrict__ batch,
                            float* __restrict__ msum, int* __restrict__ mkey,
                            float* __restrict__ cnt)
{
    int i = blockIdx.x * blockDim.x + threadIdx.x;
    if (i >= N_NODES * HID) return;
    int node = i / HID, ch = i % HID;
    int g = batch[node];
    float v = h[i];
    atomicAdd(&msum[g * HID + ch], v);
    atomicMax(&mkey[g * HID + ch], f32key(v));
    if (ch == 0) atomicAdd(&cnt[g], 1.0f);
}

__global__ void classify_kernel(const float* __restrict__ msum, const int* __restrict__ mkey,
                                const float* __restrict__ cnt,
                                const float* __restrict__ Wc1, const float* __restrict__ bc1,
                                const float* __restrict__ Wc2, const float* __restrict__ bc2,
                                float* __restrict__ out)
{
    int g = threadIdx.x;
    if (g >= N_GRAPHS) return;
    float gv[2 * HID];
    float c = cnt[g]; if (c < 1.f) c = 1.f;
    for (int k = 0; k < HID; ++k) gv[k] = msum[g * HID + k] / c;
    for (int k = 0; k < HID; ++k) {
        int kk = mkey[g * HID + k];
        gv[HID + k] = (kk == KEY_NEG_INF) ? 0.f : keyf32(kk);
    }
    float logit = bc2[0];
    for (int j = 0; j < HID; ++j) {
        float s = bc1[j];
        for (int k = 0; k < 2 * HID; ++k) s += gv[k] * Wc1[k * HID + j];
        logit += lrelu(s) * Wc2[j];
    }
    out[g] = logit;
}

// ---------------------------------------------------------------------------
extern "C" void kernel_launch(void* const* d_in, const int* in_sizes, int n_in,
                              void* d_out, int out_size, void* d_ws, size_t ws_size,
                              hipStream_t stream)
{
    const float* x     = (const float*)d_in[0];
    const int*   eidx  = (const int*)  d_in[1];
    const int*   batch = (const int*)  d_in[2];
    const float* W1    = (const float*)d_in[3];
    const float* b1    = (const float*)d_in[4];
    const float* W2    = (const float*)d_in[5];
    const float* b2    = (const float*)d_in[6];
    const float* W3    = (const float*)d_in[7];
    const float* b3    = (const float*)d_in[8];
    const float* W4    = (const float*)d_in[9];
    const float* b4    = (const float*)d_in[10];
    const float* Wc1   = (const float*)d_in[11];
    const float* bc1   = (const float*)d_in[12];
    const float* Wc2   = (const float*)d_in[13];
    const float* bc2   = (const float*)d_in[14];

    const int* src = eidx;
    const int* dst = eidx + N_EDGES;

    int*   keys = (int*)d_ws;                               // N*HID keyed max acc
    float* h1   = (float*)(keys + (size_t)N_NODES * HID);   // N*HID
    float* h2v  = h1 + (size_t)N_NODES * HID;               // N*HID
    float* msum = h2v + (size_t)N_NODES * HID;              // G*HID
    int*   mkey = (int*)(msum + N_GRAPHS * HID);            // G*HID
    float* cnt  = (float*)(mkey + N_GRAPHS * HID);          // G

    const int NH      = N_NODES * HID;
    const int fillBlk = 256;
    const int ecGrid  = 1024;

    // ---- layer 1 ----
    fill_i32<<<(NH + fillBlk - 1) / fillBlk, fillBlk, 0, stream>>>(keys, NH, KEY_NEG_INF);
    edgeconv1_wmma<<<ecGrid, 256, 0, stream>>>(x, src, dst, W1, b1, W2, b2, keys, N_EDGES);
    decode_max<<<(NH + fillBlk - 1) / fillBlk, fillBlk, 0, stream>>>(keys, h1, NH);

    // ---- layer 2 ----
    fill_i32<<<(NH + fillBlk - 1) / fillBlk, fillBlk, 0, stream>>>(keys, NH, KEY_NEG_INF);
    edgeconv2_wmma<<<ecGrid, 256, 0, stream>>>(h1, src, dst, W3, b3, W4, b4, keys, N_EDGES);
    decode_max<<<(NH + fillBlk - 1) / fillBlk, fillBlk, 0, stream>>>(keys, h2v, NH);

    // ---- pooling ----
    const int GH = N_GRAPHS * HID;
    fill_f32<<<(GH + fillBlk - 1) / fillBlk, fillBlk, 0, stream>>>(msum, GH, 0.f);
    fill_i32<<<(GH + fillBlk - 1) / fillBlk, fillBlk, 0, stream>>>(mkey, GH, KEY_NEG_INF);
    fill_f32<<<1, fillBlk, 0, stream>>>(cnt, N_GRAPHS, 0.f);
    pool_kernel<<<(NH + fillBlk - 1) / fillBlk, fillBlk, 0, stream>>>(h2v, batch, msum, mkey, cnt);

    // ---- classifier head ----
    classify_kernel<<<1, 64, 0, stream>>>(msum, mkey, cnt, Wc1, bc1, Wc2, bc2, (float*)d_out);
}
